// voltageNN_27736898798179
// MI455X (gfx1250) — matrix operand order repaired and checked by reading, not compiled, not run on hardware
//
#include <hip/hip_runtime.h>

#define HID   256
#define SEQT  1000
#define BATCH 16384
#define KPAD  1024
#define NPAD  112

typedef __attribute__((ext_vector_type(16))) _Float16 v16h;
typedef __attribute__((ext_vector_type(8)))  _Float16 v8h;
typedef __attribute__((ext_vector_type(8)))  float    v8f;

__device__ __forceinline__ float fast_sigmoid(float x) {
    // 1/(1+e^-x); large |x| saturates correctly (rcp(inf)=0)
    return __builtin_amdgcn_rcpf(1.0f + __expf(-x));
}
__device__ __forceinline__ float fast_tanh(float x) {
    // 1 - 2/(e^{2x}+1)
    return 1.0f - 2.0f * __builtin_amdgcn_rcpf(__expf(2.0f * x) + 1.0f);
}

// One wave per batch element; lane owns channels {lane + 32k, k=0..7}.
// All weights register-resident. Per-step projection = intra-wave xor reduction.
__global__ __launch_bounds__(256) void lstm_layer_kernel(
    const float* __restrict__ xin,     // (B, SEQT) input sequence (scalar per step)
    const float* __restrict__ Wih,     // (4H)  (4H x 1 flattened)
    const float* __restrict__ Whh,     // (4H)
    const float* __restrict__ bih,     // (4H)
    const float* __restrict__ bhh,     // (4H)
    const float* __restrict__ Whr,     // (H)   (1 x H flattened)
    float*       __restrict__ hout32,  // (B, SEQT) f32 output, or nullptr
    _Float16*    __restrict__ hout16)  // (B, KPAD) f16 zero-padded output, or nullptr
{
    const int lane = threadIdx.x & 31;
    const int wid  = threadIdx.x >> 5;
    const int b    = blockIdx.x * 8 + wid;

    float wxi[8], wxf[8], wxg[8], wxo[8];
    float whi[8], whf[8], whg[8], who[8];
    float bbi[8], bbf[8], bbg[8], bbo[8];
    float wr[8], c[8];
#pragma unroll
    for (int k = 0; k < 8; ++k) {
        const int ch = lane + 32 * k;
        wxi[k] = Wih[0 * HID + ch]; wxf[k] = Wih[1 * HID + ch];
        wxg[k] = Wih[2 * HID + ch]; wxo[k] = Wih[3 * HID + ch];
        whi[k] = Whh[0 * HID + ch]; whf[k] = Whh[1 * HID + ch];
        whg[k] = Whh[2 * HID + ch]; who[k] = Whh[3 * HID + ch];
        bbi[k] = bih[0 * HID + ch] + bhh[0 * HID + ch];
        bbf[k] = bih[1 * HID + ch] + bhh[1 * HID + ch];
        bbg[k] = bih[2 * HID + ch] + bhh[2 * HID + ch];
        bbo[k] = bih[3 * HID + ch] + bhh[3 * HID + ch];
        wr[k]  = Whr[ch];
        c[k]   = 0.0f;
    }

    const float* xrow = xin + (size_t)b * SEQT;
    float h = 0.0f;

    for (int t0 = 0; t0 < SEQT; t0 += 32) {
        int nsteps = SEQT - t0; if (nsteps > 32) nsteps = 32;
        // coalesced chunk load of 32 timesteps, broadcast per-step below
        float xch = (lane < nsteps) ? xrow[t0 + lane] : 0.0f;
        if (t0 + 32 + lane < SEQT)
            __builtin_prefetch(xrow + t0 + 32 + lane, 0, 0);  // global_prefetch_b8
        float hbuf = 0.0f;

        for (int s = 0; s < nsteps; ++s) {
            const float x = __shfl(xch, s, 32);
            float partial = 0.0f;
#pragma unroll
            for (int k = 0; k < 8; ++k) {
                const float gi = fmaf(x, wxi[k], fmaf(h, whi[k], bbi[k]));
                const float gf = fmaf(x, wxf[k], fmaf(h, whf[k], bbf[k]));
                const float gg = fmaf(x, wxg[k], fmaf(h, whg[k], bbg[k]));
                const float go = fmaf(x, wxo[k], fmaf(h, who[k], bbo[k]));
                const float ig = fast_sigmoid(gi);
                const float fg = fast_sigmoid(gf);
                const float gv = fast_tanh(gg);
                const float og = fast_sigmoid(go);
                c[k] = fmaf(fg, c[k], ig * gv);
                partial = fmaf(og * fast_tanh(c[k]), wr[k], partial);
            }
            // full-wave sum (projection h = sum_j o*tanh(c)*Whr[j])
#pragma unroll
            for (int m = 16; m >= 1; m >>= 1)
                partial += __shfl_xor(partial, m, 32);
            h = partial;
            if (lane == s) hbuf = h;
        }

        if (lane < nsteps) {
            if (hout32) hout32[(size_t)b * SEQT + t0 + lane] = hbuf;
            if (hout16) hout16[(size_t)b * KPAD + t0 + lane] = (_Float16)hbuf;
        }
    }
    // zero-fill K padding (cols 1000..1023) for the WMMA head
    if (hout16 && lane < (KPAD - SEQT))
        hout16[(size_t)b * KPAD + SEQT + lane] = (_Float16)0.0f;
}

// W1 (100 x 1000 f32) -> zero-padded (NPAD x KPAD) f16 B-operand buffer
__global__ __launch_bounds__(256) void convert_w1_kernel(
    const float* __restrict__ W1, _Float16* __restrict__ Bw)
{
    const int idx = blockIdx.x * 256 + threadIdx.x;  // over NPAD*KPAD
    const int n = idx >> 10;
    const int k = idx & (KPAD - 1);
    const float v = (n < 100 && k < SEQT) ? W1[n * SEQT + k] : 0.0f;
    Bw[idx] = (_Float16)v;
}

// Head: out = relu(h1 @ W1^T + b1) @ W2^T + b2, via v_wmma_f32_16x16x32_f16.
// One wave per 16-row M-tile; N padded 100->112 (7 N-tiles), K padded 1000->1024.
__global__ __launch_bounds__(256) void head_kernel(
    const _Float16* __restrict__ A,    // (B, KPAD) f16
    const _Float16* __restrict__ Bw,   // (NPAD, KPAD) f16
    const float* __restrict__ b1, const float* __restrict__ W2,
    const float* __restrict__ b2,
    float* __restrict__ out)           // (B)
{
    const int lane = threadIdx.x & 31;
    const int wid  = threadIdx.x >> 5;
    const int mt   = blockIdx.x * 8 + wid;      // M-tile id, 0..B/16-1
    const int l15  = lane & 15;
    const int hsel = lane >> 4;                 // wave half

    // A fragment: lane<16 holds row M=l15, K {0..7,16..23}; lane>=16: K {8..15,24..31}
    const _Float16* aptr  = A + (size_t)(mt * 16 + l15) * KPAD;
    const int       akoff = hsel ? 8 : 0;

    float acc[8];
#pragma unroll
    for (int j = 0; j < 8; ++j) acc[j] = 0.0f;

    for (int nt = 0; nt < 7; ++nt) {
        const int ncol = nt * 16 + l15;
        // B fragment: lane<16 holds col N=l15, K {0..7,16..23}; lane>=16: K {8..15,24..31}
        const _Float16* bptr = Bw + (size_t)ncol * KPAD + (hsel ? 8 : 0);

        v8f cacc = {};
        for (int kb = 0; kb < KPAD; kb += 32) {
            union { v16h v; v8h h[2]; } ua;
            ua.h[0] = *(const v8h*)(aptr + kb + akoff);
            ua.h[1] = *(const v8h*)(aptr + kb + 16 + akoff);
            union { v16h v; v8h h[2]; } ub;
            ub.h[0] = *(const v8h*)(bptr + kb);
            ub.h[1] = *(const v8h*)(bptr + kb + 16);
            const v16h bv = ub.v;
            cacc = __builtin_amdgcn_wmma_f32_16x16x32_f16(
                false, ua.v, false, bv, (short)0, cacc, false, false);
        }
        // C tile: VGPR j -> (M = j + 8*hsel, N = ncol). Fuse bias+ReLU+W2 weighting.
        if (ncol < 100) {
            const float bb = b1[ncol];
            const float w2 = W2[ncol];
#pragma unroll
            for (int j = 0; j < 8; ++j) {
                float v = cacc[j] + bb;
                v = v > 0.0f ? v : 0.0f;
                acc[j] = fmaf(v, w2, acc[j]);
            }
        }
    }
    // reduce over N: sum across the 16 lanes of each half (rows differ across halves)
#pragma unroll
    for (int m = 8; m >= 1; m >>= 1) {
#pragma unroll
        for (int j = 0; j < 8; ++j)
            acc[j] += __shfl_xor(acc[j], m, 32);
    }
    if (l15 == 0) {
        const float bias2 = b2[0];
#pragma unroll
        for (int j = 0; j < 8; ++j)
            out[mt * 16 + hsel * 8 + j] = acc[j] + bias2;
    }
}

extern "C" void kernel_launch(void* const* d_in, const int* in_sizes, int n_in,
                              void* d_out, int out_size, void* d_ws, size_t ws_size,
                              hipStream_t stream) {
    (void)in_sizes; (void)n_in; (void)out_size; (void)ws_size;
    const float* x    = (const float*)d_in[0];
    const float* Wih0 = (const float*)d_in[1];
    const float* Whh0 = (const float*)d_in[2];
    const float* bih0 = (const float*)d_in[3];
    const float* bhh0 = (const float*)d_in[4];
    const float* Whr0 = (const float*)d_in[5];
    const float* Wih1 = (const float*)d_in[6];
    const float* Whh1 = (const float*)d_in[7];
    const float* bih1 = (const float*)d_in[8];
    const float* bhh1 = (const float*)d_in[9];
    const float* Whr1 = (const float*)d_in[10];
    const float* W1   = (const float*)d_in[11];
    const float* b1   = (const float*)d_in[12];
    const float* W2   = (const float*)d_in[13];
    const float* b2   = (const float*)d_in[14];
    float* out = (float*)d_out;

    char* ws = (char*)d_ws;
    const size_t h0_bytes = (size_t)BATCH * SEQT * sizeof(float);       // 65.5 MB
    const size_t h1_bytes = (size_t)BATCH * KPAD * sizeof(_Float16);    // 33.6 MB
    float*    h0  = (float*)ws;
    _Float16* h1  = (_Float16*)(ws + h0_bytes);
    _Float16* w1h = (_Float16*)(ws + h0_bytes + h1_bytes);              // +229 KB

    convert_w1_kernel<<<(NPAD * KPAD) / 256, 256, 0, stream>>>(W1, w1h);
    lstm_layer_kernel<<<BATCH / 8, 256, 0, stream>>>(
        x, Wih0, Whh0, bih0, bhh0, Whr0, h0, (_Float16*)nullptr);
    lstm_layer_kernel<<<BATCH / 8, 256, 0, stream>>>(
        h0, Wih1, Whh1, bih1, bhh1, Whr1, (float*)nullptr, h1);
    head_kernel<<<BATCH / (16 * 8), 256, 0, stream>>>(h1, w1h, b1, W2, b2, out);
}
